// MPNN_4071628997308
// MI455X (gfx1250) — compile-verified, matrix operand order checked by reading, not run on hardware
//
#include <hip/hip_runtime.h>

// ---------------------------------------------------------------------------
// MPNN pipeline for gfx1250 (MI455X): bf16 WMMA (16x16x32, f32 accum) for all
// GEMM stages, LDS-staged tiles, wave32 / 8 waves per 256-thread block.
// Weights are pre-packed (bf16, transposed) once per GEMM call so the 16K
// GEMM blocks do only coalesced b128 copies. Scores tiles use gfx1250 async
// global->LDS copies (ASYNCcnt) — builtin presence probe-confirmed this round.
// ---------------------------------------------------------------------------

#define M_   128
#define N_   64
#define MN_  (M_ * N_)        // 8192
#define R_   (M_ * N_ * N_)   // 524288 pair rows
#define LDY  96               // padded Yij feature stride (bf16)

#define USE_ASYNC_LDS 0
#if defined(__has_builtin)
#if __has_builtin(__builtin_amdgcn_global_load_async_to_lds_b128) && \
    __has_builtin(__builtin_amdgcn_s_wait_asynccnt)
#undef USE_ASYNC_LDS
#define USE_ASYNC_LDS 1
#endif
#endif

typedef __bf16 bf16;
typedef bf16  v16bf __attribute__((ext_vector_type(16)));
typedef float v8f   __attribute__((ext_vector_type(8)));
typedef int   v4i   __attribute__((ext_vector_type(4)));

union FragAB { v16bf v; uint4 q[2]; };
union AccF   { v8f v; float a[8]; };

__device__ __forceinline__ unsigned short f2b(float f) {
  union { float f; unsigned u; } v; v.f = f;
  unsigned r = v.u + 0x7FFFu + ((v.u >> 16) & 1u);   // round-to-nearest-even
  return (unsigned short)(r >> 16);
}
__device__ __forceinline__ float b2f(unsigned short h) {
  union { unsigned u; float f; } v; v.u = ((unsigned)h) << 16;
  return v.f;
}
__device__ __forceinline__ float gelu_t(float x) {   // jax.nn.gelu (tanh approx)
  return 0.5f * x * (1.0f + tanhf(0.7978845608028654f * (x + 0.044715f * x * x * x)));
}

// Load a 16x32 bf16 operand fragment (A-style) from an LDS row-major tile.
// ISA layout: lanes 0-15 row=lane, K={0..7,16..23}; lanes 16-31 row=lane&15,
// K={8..15,24..31}. Two 16-byte LDS loads per lane. ld must be a multiple of 8.
__device__ __forceinline__ v16bf ld_frag(const unsigned short* base, int ld) {
  unsigned l  = threadIdx.x & 31u;
  unsigned r  = l & 15u;
  unsigned k0 = (l < 16u) ? 0u : 8u;
  const unsigned short* p = base + (size_t)r * (unsigned)ld + k0;
  FragAB f;
  f.q[0] = *(const uint4*)(p);
  f.q[1] = *(const uint4*)(p + 16);
  return f.v;
}

__device__ __forceinline__ v8f wmma_bf16(v16bf a, v16bf b, v8f c) {
  // (neg_a, A, neg_b, B, c_mod, C, reuse_a, reuse_b)
  return __builtin_amdgcn_wmma_f32_16x16x32_bf16(false, a, false, b, (short)0, c,
                                                 false, false);
}

// ---------------------------------------------------------------------------
// Stage 0: pair features Iij (f32) + bf16 Yij buffers (stride LDY, zero-padded)
// ---------------------------------------------------------------------------
__global__ void __launch_bounds__(256) setup_kernel(
    const float* __restrict__ x, const float* __restrict__ embed,
    float* __restrict__ Iij, unsigned short* __restrict__ YijA,
    unsigned short* __restrict__ YijB)
{
  int row = blockIdx.x * 256 + threadIdx.x;
  if (row >= R_) return;
  int m = row >> 12, i = (row >> 6) & 63, j = row & 63;
  float r0 = x[(m * 64 + i) * 2 + 0] - x[(m * 64 + j) * 2 + 0];
  float r1 = x[(m * 64 + i) * 2 + 1] - x[(m * 64 + j) * 2 + 1];
  const float c1 = 0.6283185307179586f;   // 2*pi/L
  const float c2 = 0.3141592653589793f;   // pi/L
  float eye = (i == j) ? 1.0f : 0.0f;
  float s0 = sinf(c2 * r0) + eye, s1 = sinf(c2 * r1) + eye;
  float dsin = sqrtf(s0 * s0 + s1 * s1) * (1.0f - eye);
  float ss = ((i < 32) == (j < 32)) ? 1.0f : -1.0f;
  float v[6] = { sinf(c1 * r0), sinf(c1 * r1), cosf(c1 * r0), cosf(c1 * r1), dsin, ss };
  size_t rb = (size_t)row * LDY;
#pragma unroll
  for (int c = 0; c < 6; ++c) {
    Iij[(size_t)row * 6 + c] = v[c];
    unsigned short h = f2b(v[c]);
    YijA[rb + c] = h;
    YijB[rb + c] = h;
  }
  for (int c = 0; c < 32; ++c) YijA[rb + 6 + c] = f2b(embed[32 + c]);  // emb[1]
  for (int c = 38; c < LDY; ++c) YijA[rb + c] = 0;
  for (int c = 6; c < LDY; ++c)  YijB[rb + c] = 0;
}

__global__ void __launch_bounds__(256) yi_init_kernel(
    const float* __restrict__ embed, float* __restrict__ yi)
{
  int idx = blockIdx.x * 256 + threadIdx.x;
  if (idx >= MN_ * 64) return;
  int c = idx & 63;
  yi[idx] = (c < 32) ? embed[c] : 0.0f;   // emb[0], zero-padded to 64
}

// ---------------------------------------------------------------------------
// Weight packers (single block): transpose + f32->bf16 once per GEMM call so
// GEMM blocks only do coalesced b128 LDS fills.
// ---------------------------------------------------------------------------
__global__ void __launch_bounds__(256) pack_qk_kernel(
    const float* __restrict__ wq, const float* __restrict__ wk,
    int din_real, int din_pad, unsigned short* __restrict__ wbuf)
{
  int tid = threadIdx.x;
  for (int idx = tid; idx < 64 * din_pad; idx += 256) {
    int c = idx / din_pad, k = idx - c * din_pad;
    float v = 0.0f;
    if (k < din_real) v = (c < 32) ? wq[k * 32 + c] : wk[k * 32 + (c - 32)];
    wbuf[idx] = f2b(v);
  }
}

// Packed MLP weight layout: [64*din_pad] W1T bf16 | [4096] W2T bf16 |
//                           [64] b1 f32 | [64] b2 f32
__global__ void __launch_bounds__(256) pack_mlp_kernel(
    const float* __restrict__ W1, const float* __restrict__ B1,
    const float* __restrict__ W2, const float* __restrict__ B2,
    int din_real, int din_pad, unsigned short* __restrict__ wbuf)
{
  int tid = threadIdx.x;
  for (int idx = tid; idx < 64 * din_pad; idx += 256) {
    int h = idx / din_pad, k = idx - h * din_pad;
    wbuf[idx] = f2b((k < din_real) ? W1[k * 64 + h] : 0.0f);
  }
  unsigned short* w2t = wbuf + 64 * din_pad;
  for (int idx = tid; idx < 4096; idx += 256) {
    int h = idx >> 6, k = idx & 63;
    w2t[idx] = f2b(W2[k * 64 + h]);
  }
  float* bias = (float*)(w2t + 4096);
  if (tid < 64) { bias[tid] = B1[tid]; bias[64 + tid] = B2[tid]; }
}

// ---------------------------------------------------------------------------
// Stage 1: xq|xk = Yij @ [wq|wk]   (rowwise GEMM, dout = 64 = 32+32)
// Epilogue scatters xq -> [m][k][i][j], xk -> [m][k][l][j] (pre-transposed B).
// ---------------------------------------------------------------------------
__global__ void __launch_bounds__(256) qk_kernel(
    const unsigned short* __restrict__ Yij, int din_pad,
    const unsigned short* __restrict__ wbuf,
    unsigned short* __restrict__ xq, unsigned short* __restrict__ xk)
{
  __shared__ __align__(16) unsigned short WT[64 * 96];
  __shared__ __align__(16) unsigned short Xs[32 * 96];
  int tid = threadIdx.x, wave = tid >> 5;
  int row0 = blockIdx.x * 32;

  {
    const uint4* src = (const uint4*)wbuf;
    uint4* dst = (uint4*)WT;
    int n = (64 * din_pad) >> 3;
    for (int idx = tid; idx < n; idx += 256) dst[idx] = src[idx];
  }
  for (int idx = tid; idx < 32 * din_pad; idx += 256) {
    int r = idx / din_pad, c = idx - r * din_pad;
    Xs[r * din_pad + c] = Yij[(size_t)(row0 + r) * LDY + c];
  }
  __syncthreads();

  int r0 = (wave & 1) * 16, c0 = (wave >> 1) * 16;
  AccF acc;
#pragma unroll
  for (int i = 0; i < 8; ++i) acc.a[i] = 0.0f;
  for (int kk = 0; kk < din_pad; kk += 32) {
    v16bf a = ld_frag(Xs + r0 * din_pad + kk, din_pad);
    v16bf b = ld_frag(WT + c0 * din_pad + kk, din_pad);
    acc.v = wmma_bf16(a, b, acc.v);
  }
  unsigned l = tid & 31u;
  int col = c0 + (int)(l & 15u);
  int rb  = r0 + ((l < 16u) ? 0 : 8);
#pragma unroll
  for (int v = 0; v < 8; ++v) {
    int row = row0 + rb + v;
    int m = row >> 12, aI = (row >> 6) & 63, bJ = row & 63;
    unsigned short h = f2b(acc.a[v]);
    if (col < 32) xq[((size_t)(m * 32 + col) << 12) + aI * 64 + bJ] = h;
    else          xk[((size_t)(m * 32 + (col - 32)) << 12) + bJ * 64 + aI] = h;
  }
}

// ---------------------------------------------------------------------------
// Stage 2: scores[m,i,l,k] = (1/sqrt(A)) * Sum_j xq[m,k,i,j] * xkT[m,k,l,j]
// 4096 blocks, each one 64x64x64 bf16 GEMM entirely in LDS.
// Tile staging uses gfx1250 async global->LDS copies (ASYNCcnt tracked).
// ---------------------------------------------------------------------------
__global__ void __launch_bounds__(256) scores_kernel(
    const unsigned short* __restrict__ xq, const unsigned short* __restrict__ xk,
    unsigned short* __restrict__ scores)
{
  __shared__ __align__(16) unsigned short As[64 * 64];
  __shared__ __align__(16) unsigned short Bs[64 * 64];
  int tid = threadIdx.x, wave = tid >> 5;
  int mk = blockIdx.x;                  // m*32 + k
  size_t base = (size_t)mk << 12;       // * 4096 elements
  const uint4* aSrc = (const uint4*)(xq + base);
  const uint4* bSrc = (const uint4*)(xk + base);
  uint4* aDst = (uint4*)As;
  uint4* bDst = (uint4*)Bs;
#if USE_ASYNC_LDS
  for (int idx = tid; idx < 512; idx += 256) {
    __builtin_amdgcn_global_load_async_to_lds_b128(
        (v4i*)(aSrc + idx), (v4i*)(aDst + idx), 0, 0);
    __builtin_amdgcn_global_load_async_to_lds_b128(
        (v4i*)(bSrc + idx), (v4i*)(bDst + idx), 0, 0);
  }
  __builtin_amdgcn_s_wait_asynccnt(0);
#else
  for (int idx = tid; idx < 512; idx += 256) { aDst[idx] = aSrc[idx]; bDst[idx] = bSrc[idx]; }
#endif
  __syncthreads();

  int m = mk >> 5, k = mk & 31;
  const float scale = 0.17677669529663688f;   // 1/sqrt(32)
  for (int t = wave; t < 16; t += 8) {
    int ti = t >> 2, tj = t & 3;
    AccF acc;
#pragma unroll
    for (int i = 0; i < 8; ++i) acc.a[i] = 0.0f;
#pragma unroll
    for (int kk = 0; kk < 64; kk += 32) {
      v16bf a = ld_frag(As + ti * 16 * 64 + kk, 64);
      v16bf b = ld_frag(Bs + tj * 16 * 64 + kk, 64);
      acc.v = wmma_bf16(a, b, acc.v);
    }
    unsigned l = tid & 31u;
    int col = tj * 16 + (int)(l & 15u);
    int rb  = ti * 16 + ((l < 16u) ? 0 : 8);
#pragma unroll
    for (int v = 0; v < 8; ++v) {
      int i = rb + v;
      scores[((size_t)m * 4096 + (size_t)i * 64 + col) * 32 + k] = f2b(acc.a[v] * scale);
    }
  }
}

// ---------------------------------------------------------------------------
// Generic fused 2-layer MLP: Y = gelu(X @ W1 + b1) @ W2 + b2, rowwise.
// X = [X1 cols 0..d1) | X2 cols d1..d1+d2) | zero pad to din_pad].
// Weights come pre-packed (bf16, transposed) from wbuf.
// Optional epilogues: elementwise multiply by mulsrc (bf16, ld 64),
// bf16 store into outB at column offset coff (ld ldo), f32 store into outF.
// ---------------------------------------------------------------------------
__global__ void __launch_bounds__(256) fused_mlp_kernel(
    const unsigned short* __restrict__ X1, int ld1, int d1,
    const unsigned short* __restrict__ X2, int ld2, int d2,
    int din_pad,
    const unsigned short* __restrict__ wbuf,
    const unsigned short* __restrict__ mulsrc,
    unsigned short* __restrict__ outB, int ldo, int coff,
    float* __restrict__ outF)
{
  __shared__ __align__(16) unsigned short W1T[64 * 160];
  __shared__ __align__(16) unsigned short W2T[64 * 64];
  __shared__ __align__(16) unsigned short Xs[32 * 160];
  __shared__ __align__(16) unsigned short Hs[32 * 64];
  __shared__ float b1s[64];
  __shared__ float b2s[64];

  int tid = threadIdx.x, wave = tid >> 5;
  int row0 = blockIdx.x * 32;
  int din = d1 + d2;

  {   // coalesced b128 copies of packed weights into LDS
    const uint4* src = (const uint4*)wbuf;
    uint4* dst = (uint4*)W1T;
    int n1 = (64 * din_pad) >> 3;
    for (int idx = tid; idx < n1; idx += 256) dst[idx] = src[idx];
    const uint4* src2 = (const uint4*)(wbuf + 64 * din_pad);
    uint4* dst2 = (uint4*)W2T;
    for (int idx = tid; idx < 512; idx += 256) dst2[idx] = src2[idx];
    const float* bias = (const float*)(wbuf + 64 * din_pad + 4096);
    if (tid < 64) { b1s[tid] = bias[tid]; b2s[tid] = bias[64 + tid]; }
  }
  for (int idx = tid; idx < 32 * din_pad; idx += 256) {
    int r = idx / din_pad, c = idx - r * din_pad;
    size_t row = (size_t)(row0 + r);
    unsigned short v = 0;
    if (c < d1)       v = X1[row * (unsigned)ld1 + c];
    else if (c < din) v = X2[row * (unsigned)ld2 + (c - d1)];
    Xs[r * din_pad + c] = v;
  }
  if (tid < 32) {
    // speculative prefetch of next row-block into GL2 (global_prefetch_b8)
    __builtin_prefetch(X1 + ((size_t)row0 + 32 + tid) * (unsigned)ld1, 0, 1);
  }
  __syncthreads();

  int r0 = (wave & 1) * 16, c0 = (wave >> 1) * 16;
  unsigned l = tid & 31u;
  int col = (int)(l & 15u);
  int rb  = (l < 16u) ? 0 : 8;

  {   // hidden layer: H = gelu(X @ W1 + b1)
    AccF acc;
#pragma unroll
    for (int i = 0; i < 8; ++i) acc.a[i] = 0.0f;
    for (int kk = 0; kk < din_pad; kk += 32) {
      v16bf a = ld_frag(Xs + r0 * din_pad + kk, din_pad);
      v16bf b = ld_frag(W1T + c0 * din_pad + kk, din_pad);
      acc.v = wmma_bf16(a, b, acc.v);
    }
    float bias = b1s[c0 + col];
#pragma unroll
    for (int v = 0; v < 8; ++v)
      Hs[(r0 + rb + v) * 64 + c0 + col] = f2b(gelu_t(acc.a[v] + bias));
  }
  __syncthreads();
  {   // output layer: Y = H @ W2 + b2 (+ epilogue)
    AccF acc;
#pragma unroll
    for (int i = 0; i < 8; ++i) acc.a[i] = 0.0f;
#pragma unroll
    for (int kk = 0; kk < 64; kk += 32) {
      v16bf a = ld_frag(Hs + r0 * 64 + kk, 64);
      v16bf b = ld_frag(W2T + c0 * 64 + kk, 64);
      acc.v = wmma_bf16(a, b, acc.v);
    }
    float bias = b2s[c0 + col];
#pragma unroll
    for (int v = 0; v < 8; ++v) {
      size_t row = (size_t)(row0 + r0 + rb + v);
      float val = acc.a[v] + bias;
      if (mulsrc) val *= b2f(mulsrc[row * 64 + c0 + col]);
      if (outB)   outB[row * (unsigned)ldo + coff + c0 + col] = f2b(val);
      if (outF)   outF[row * 64 + c0 + col] = val;
    }
  }
}

// ---------------------------------------------------------------------------
// agg[m,i,:] = Sum_j Mij[m,i,j,:]; build X3 = [yi(bf16) | agg | zero pad 128].
// ---------------------------------------------------------------------------
__global__ void __launch_bounds__(256) agg_concat_kernel(
    const unsigned short* __restrict__ Mij, const float* __restrict__ yi,
    int dyi, unsigned short* __restrict__ X3)
{
  int t = threadIdx.x;
  int row = blockIdx.x * 4 + (t >> 6);   // (m*64 + i), 8192 total
  int e = t & 63;
  const unsigned short* p = Mij + ((size_t)row << 12) + e;
  float s = 0.0f;
  for (int j = 0; j < 64; ++j) s += b2f(p[j * 64]);
  X3[(size_t)row * 128 + dyi + e] = f2b(s);
  if (e < dyi) X3[(size_t)row * 128 + e] = f2b(yi[(size_t)row * 64 + e]);
  for (int c = dyi + 64 + e; c < 128; c += 64) X3[(size_t)row * 128 + c] = 0;
}

// ---------------------------------------------------------------------------
// Emit f32 outputs: yi (MN*64) then Yij = [Iij(6) | mlp4_out(64)] per pair row.
// ---------------------------------------------------------------------------
__global__ void __launch_bounds__(256) emit_kernel(
    const float* __restrict__ yi, const float* __restrict__ Iij,
    const unsigned short* __restrict__ Yf, float* __restrict__ out)
{
  size_t idx = (size_t)blockIdx.x * 256 + threadIdx.x;
  const size_t nyi = (size_t)MN_ * 64;
  const size_t tot = nyi + (size_t)R_ * 70;
  if (idx >= tot) return;
  if (idx < nyi) { out[idx] = yi[idx]; return; }
  size_t k = idx - nyi;
  size_t row = k / 70;
  int c = (int)(k - row * 70);
  out[idx] = (c < 6) ? Iij[row * 6 + c] : b2f(Yf[row * LDY + c]);
}

// ---------------------------------------------------------------------------
// Host-side orchestration.
// d_in order (setup_inputs insertion order): x, embed, then per layer:
//   wq, wk, mlp1{l1W,l1b,l2W,l2b}, mlp2{...}, mlp3{...}, mlp4{...}  (18 each)
// ---------------------------------------------------------------------------
extern "C" void kernel_launch(void* const* d_in, const int* in_sizes, int n_in,
                              void* d_out, int out_size, void* d_ws, size_t ws_size,
                              hipStream_t stream)
{
  (void)in_sizes; (void)n_in; (void)out_size; (void)ws_size;
  const float* x     = (const float*)d_in[0];
  const float* embed = (const float*)d_in[1];

  char* ws = (char*)d_ws;
  size_t off = 0;
  auto alloc = [&](size_t bytes) -> void* {
    void* p = ws + off;
    off += (bytes + 255) & ~(size_t)255;
    return p;
  };
  float*          Iij  = (float*)alloc((size_t)R_ * 6 * 4);
  unsigned short* YijA = (unsigned short*)alloc((size_t)R_ * LDY * 2);
  unsigned short* YijB = (unsigned short*)alloc((size_t)R_ * LDY * 2);
  unsigned short* xq   = (unsigned short*)alloc((size_t)M_ * 32 * 4096 * 2);
  unsigned short* xk   = (unsigned short*)alloc((size_t)M_ * 32 * 4096 * 2);
  unsigned short* sc   = (unsigned short*)alloc((size_t)R_ * 32 * 2);
  unsigned short* att  = (unsigned short*)alloc((size_t)R_ * 64 * 2);
  unsigned short* mij  = (unsigned short*)alloc((size_t)R_ * 64 * 2);
  unsigned short* X3   = (unsigned short*)alloc((size_t)MN_ * 128 * 2);
  float*          yi   = (float*)alloc((size_t)MN_ * 64 * 4);
  unsigned short* wqk  = (unsigned short*)alloc(64 * 96 * 2);
  unsigned short* wmlp = (unsigned short*)alloc(64 * 160 * 2 + 4096 * 2 + 512);

  setup_kernel<<<R_ / 256, 256, 0, stream>>>(x, embed, Iij, YijA, YijB);
  yi_init_kernel<<<(MN_ * 64) / 256, 256, 0, stream>>>(embed, yi);

  for (int lp = 0; lp < 2; ++lp) {
    int base = 2 + lp * 18;
    const float* wq_  = (const float*)d_in[base + 0];
    const float* wk_  = (const float*)d_in[base + 1];
    const float* m1W1 = (const float*)d_in[base + 2];
    const float* m1b1 = (const float*)d_in[base + 3];
    const float* m1W2 = (const float*)d_in[base + 4];
    const float* m1b2 = (const float*)d_in[base + 5];
    const float* m2W1 = (const float*)d_in[base + 6];
    const float* m2b1 = (const float*)d_in[base + 7];
    const float* m2W2 = (const float*)d_in[base + 8];
    const float* m2b2 = (const float*)d_in[base + 9];
    const float* m3W1 = (const float*)d_in[base + 10];
    const float* m3b1 = (const float*)d_in[base + 11];
    const float* m3W2 = (const float*)d_in[base + 12];
    const float* m3b2 = (const float*)d_in[base + 13];
    const float* m4W1 = (const float*)d_in[base + 14];
    const float* m4b1 = (const float*)d_in[base + 15];
    const float* m4W2 = (const float*)d_in[base + 16];
    const float* m4b2 = (const float*)d_in[base + 17];

    int yij_dim = (lp == 0) ? 38 : 70;
    int dpy     = (lp == 0) ? 64 : 96;      // Yij K padded to 32-multiple
    int dyi     = (lp == 0) ? 32 : 64;      // yi width feeding mlp3
    unsigned short* cur = (lp == 0) ? YijA : YijB;
    unsigned short* nxt = (lp == 0) ? YijB : YijA;

    pack_qk_kernel<<<1, 256, 0, stream>>>(wq_, wk_, yij_dim, dpy, wqk);
    qk_kernel<<<R_ / 32, 256, 0, stream>>>(cur, dpy, wqk, xq, xk);
    scores_kernel<<<M_ * 32, 256, 0, stream>>>(xq, xk, sc);

    // attn = mlp1(scores)
    pack_mlp_kernel<<<1, 256, 0, stream>>>(m1W1, m1b1, m1W2, m1b2, 32, 32, wmlp);
    fused_mlp_kernel<<<R_ / 32, 256, 0, stream>>>(
        sc, 32, 32, nullptr, 0, 0, 32, wmlp, nullptr, att, 64, 0, nullptr);

    // Mij = attn * mlp2(Yij)
    pack_mlp_kernel<<<1, 256, 0, stream>>>(m2W1, m2b1, m2W2, m2b2, yij_dim, dpy, wmlp);
    fused_mlp_kernel<<<R_ / 32, 256, 0, stream>>>(
        cur, LDY, yij_dim, nullptr, 0, 0, dpy, wmlp, att, mij, 64, 0, nullptr);

    // agg + concat(yi, agg)
    agg_concat_kernel<<<MN_ / 4, 256, 0, stream>>>(mij, yi, dyi, X3);

    // yi = mlp3(X3)
    int d3p = (dyi + 64 + 31) & ~31;
    pack_mlp_kernel<<<1, 256, 0, stream>>>(m3W1, m3b1, m3W2, m3b2, dyi + 64, d3p, wmlp);
    fused_mlp_kernel<<<MN_ / 32, 256, 0, stream>>>(
        X3, 128, dyi + 64, nullptr, 0, 0, d3p, wmlp, nullptr, nullptr, 0, 0, yi);

    // Yij_next[:,6:70] = mlp4(concat(Yij, Mij))
    int d4p = (yij_dim + 64 + 31) & ~31;
    pack_mlp_kernel<<<1, 256, 0, stream>>>(m4W1, m4b1, m4W2, m4b2, yij_dim + 64, d4p, wmlp);
    fused_mlp_kernel<<<R_ / 32, 256, 0, stream>>>(
        cur, LDY, yij_dim, mij, 64, 64, d4p, wmlp, nullptr, nxt, LDY, 6, nullptr);
  }

  size_t tot = (size_t)MN_ * 64 + (size_t)R_ * 70;
  emit_kernel<<<(int)((tot + 255) / 256), 256, 0, stream>>>(yi, Iij, YijA, (float*)d_out);
}
